// AttentionLayer_28913719837307
// MI455X (gfx1250) — matrix operand order repaired and checked by reading, not compiled, not run on hardware
//
#include <hip/hip_runtime.h>

#define Bn  32
#define SDn 1024
#define SEn 1024
#define Dn  256

typedef __attribute__((ext_vector_type(16))) __bf16 v16bf;
typedef __attribute__((ext_vector_type(8)))  __bf16 v8bf;
typedef __attribute__((ext_vector_type(8)))  float  v8f;

// round-to-nearest-even f32 -> bf16 without relying on cast support
__device__ __forceinline__ __bf16 f2bf(float x) {
  unsigned u = __builtin_bit_cast(unsigned, x);
  u += 0x7FFFu + ((u >> 16) & 1u);
  unsigned short s = (unsigned short)(u >> 16);
  return __builtin_bit_cast(__bf16, s);
}

// A-fragment (16x32 bf16, ISA 7.12.2): lane-half h holds K in [k0+8h,+8) and [k0+16+8h,+8)
__device__ __forceinline__ v16bf ldA(const __bf16* rowp, int k0, int h) {
  v8bf lo = *(const v8bf*)(rowp + k0 + 8 * h);
  v8bf hi = *(const v8bf*)(rowp + k0 + 16 + 8 * h);
  v16bf f;
#pragma unroll
  for (int i = 0; i < 8; ++i) { f[i] = lo[i]; f[i + 8] = hi[i]; }
  return f;
}

// B-fragment (32x16 bf16): lane-half h holds K in [k0+8h,+8) and [k0+16+8h,+8)
__device__ __forceinline__ v16bf ldB(const __bf16* colp, int k0, int h) {
  v8bf lo = *(const v8bf*)(colp + k0 + 8 * h);
  v8bf hi = *(const v8bf*)(colp + k0 + 16 + 8 * h);
  v16bf f;
#pragma unroll
  for (int i = 0; i < 8; ++i) { f[i] = lo[i]; f[i + 8] = hi[i]; }
  return f;
}

__device__ __forceinline__ v8f wmma_bf16(v16bf a, v16bf b, v8f c) {
  return __builtin_amdgcn_wmma_f32_16x16x32_bf16(false, a, false, b, (short)0, c,
                                                 false, false);
}

// ---------------- kernel 0a: f32 -> bf16 for padded_seqs and W_attn ----------------
__global__ void cvt_pw_kernel(const float* __restrict__ p, const float* __restrict__ W,
                              __bf16* __restrict__ pbf, __bf16* __restrict__ wbf) {
  size_t idx = (size_t)blockIdx.x * blockDim.x + threadIdx.x;
  size_t stride = (size_t)gridDim.x * blockDim.x;
  for (size_t i = idx; i < (size_t)Bn * SDn * Dn; i += stride) pbf[i] = f2bf(p[i]);
  for (size_t i = idx; i < (size_t)Dn * 2 * Dn; i += stride) wbf[i] = f2bf(W[i]);
}

// -------- kernel 0b: encoder f32 -> bf16 ([e][d]) AND transposed bf16 ([d][e]) --------
__global__ void enc_cvt_tr_kernel(const float* __restrict__ e,
                                  __bf16* __restrict__ ebf, __bf16* __restrict__ etbf) {
  __shared__ float tile[32][33];
  const int b = blockIdx.z, e0 = blockIdx.x * 32, d0 = blockIdx.y * 32;
  const int tid = threadIdx.x;
  const int li = tid >> 3, lj = (tid & 7) * 4;
  const float* src = e + ((size_t)b * SEn + e0 + li) * Dn + d0 + lj;
  float4 v = *(const float4*)src;
  __bf16* eb = ebf + ((size_t)b * SEn + e0 + li) * Dn + d0 + lj;
  eb[0] = f2bf(v.x); eb[1] = f2bf(v.y); eb[2] = f2bf(v.z); eb[3] = f2bf(v.w);
  tile[li][lj + 0] = v.x; tile[li][lj + 1] = v.y;
  tile[li][lj + 2] = v.z; tile[li][lj + 3] = v.w;
  __syncthreads();
  __bf16* et = etbf + ((size_t)b * Dn + d0 + li) * SEn + e0 + lj;
#pragma unroll
  for (int k = 0; k < 4; ++k) et[k] = f2bf(tile[lj + k][li]);
}

// ---- kernel 1: scores (WMMA) + softmax + weights out + context GEMM (WMMA) ----
// one workgroup (8 wave32) = one (batch, 32-q-row) tile
// LDS: 128 KiB f32 scores | 64 KiB bf16 weights | 1 KiB reduction
__global__ void attn_kernel(const __bf16* __restrict__ pbf, const __bf16* __restrict__ ebf,
                            const __bf16* __restrict__ etbf, float* __restrict__ wout,
                            __bf16* __restrict__ ctxbf) {
  extern __shared__ __align__(16) char smem_raw[];
  float*  sc  = (float*)smem_raw;                        // 32 x 1024 f32
  __bf16* wb  = (__bf16*)(smem_raw + 32 * SEn * 4);      // 32 x 1024 bf16
  float*  red = (float*)(smem_raw + 32 * SEn * 4 + 32 * SEn * 2); // 32 x 8 f32

  const int b = blockIdx.y, q0 = blockIdx.x * 32;
  const int tid = threadIdx.x;
  const int w = tid >> 5, l = tid & 31;
  const int r = l & 15, h = l >> 4;
  const int qs = w & 1;

  // ---- scores = Q . E^T, contraction over d (K=256 -> 8 chunks of 32) ----
  const __bf16* Pa = pbf + ((size_t)b * SDn + q0 + qs * 16 + r) * Dn;
  const __bf16* Eb = ebf + (size_t)b * SEn * Dn;

  v16bf Af[8];
#pragma unroll
  for (int c = 0; c < 8; ++c) Af[c] = ldA(Pa, c * 32, h);

  for (int t = 0; t < 16; ++t) {
    const int e0 = ((w >> 1) + 4 * t) * 16;
    const __bf16* Bc = Eb + (size_t)(e0 + r) * Dn;
    if (t + 1 < 16) __builtin_prefetch(Eb + (size_t)(e0 + 64 + r) * Dn, 0, 3);
    v8f acc = {};
#pragma unroll
    for (int c = 0; c < 8; ++c) acc = wmma_bf16(Af[c], ldB(Bc, c * 32, h), acc);
#pragma unroll
    for (int i = 0; i < 8; ++i)
      sc[(qs * 16 + h * 8 + i) * SEn + e0 + r] = acc[i] * 0.0625f; // 1/sqrt(256)
  }
  __syncthreads();

  // ---- softmax across SE per row; also emit bf16 weights into LDS ----
  {
    const int row = tid >> 3, sl = tid & 7;
    float* srow = sc + row * SEn;
    __bf16* wrow = wb + row * SEn;
    float mx = -3.402823466e38f;
    for (int e2 = sl; e2 < SEn; e2 += 8) mx = fmaxf(mx, srow[e2]);
    red[row * 8 + sl] = mx;
    __syncthreads();
#pragma unroll
    for (int j = 0; j < 8; ++j) mx = fmaxf(mx, red[row * 8 + j]);
    float sum = 0.f;
    for (int e2 = sl; e2 < SEn; e2 += 8) {
      float v = __expf(srow[e2] - mx);
      srow[e2] = v;
      sum += v;
    }
    __syncthreads();
    red[row * 8 + sl] = sum;
    __syncthreads();
    sum = 0.f;
#pragma unroll
    for (int j = 0; j < 8; ++j) sum += red[row * 8 + j];
    const float inv = 1.0f / sum;
    for (int e2 = sl; e2 < SEn; e2 += 8) {
      float v = srow[e2] * inv;
      srow[e2] = v;          // f32 copy for global weights output
      wrow[e2] = f2bf(v);    // bf16 copy = A operand of context GEMM
    }
  }
  __syncthreads();

  // ---- write attention weights, coalesced float4 ----
  {
    float4* g4 = (float4*)(wout + ((size_t)b * SDn + q0) * SEn);
    const float4* s4 = (const float4*)sc;
    for (int i = tid; i < 32 * SEn / 4; i += 256) g4[i] = s4[i];
  }

  // ---- context = weights(32x1024) @ E(1024x256); B-fragments from E^T ----
  // chunk-outer / tile-inner: 1 LDS A-fragment feeds 4 independent WMMAs
  const __bf16* ET = etbf + (size_t)b * Dn * SEn;
  const __bf16* Aw = wb + (qs * 16 + r) * SEn;
  const __bf16* Bcs[4];
  int d0s[4];
#pragma unroll
  for (int t = 0; t < 4; ++t) {
    d0s[t] = ((w >> 1) + 4 * t) * 16;
    Bcs[t] = ET + (size_t)(d0s[t] + r) * SEn;
  }
  v8f acc[4] = {};
  for (int c = 0; c < 32; ++c) { // K = 1024 -> 32 chunks of 32
    v16bf Afr = ldA(Aw, c * 32, h);
#pragma unroll
    for (int t = 0; t < 4; ++t)
      acc[t] = wmma_bf16(Afr, ldB(Bcs[t], c * 32, h), acc[t]);
  }
  __bf16* cr = ctxbf + ((size_t)b * SDn + q0 + qs * 16) * Dn;
#pragma unroll
  for (int t = 0; t < 4; ++t)
#pragma unroll
    for (int i = 0; i < 8; ++i)
      cr[(size_t)(h * 8 + i) * Dn + d0s[t] + r] = f2bf(acc[t][i]);
}

// ---- kernel 2: logits = tanh([p, ctx] @ W^T + b) * mask  (M=32768,N=256,K=512) ----
__global__ void proj_kernel(const __bf16* __restrict__ pbf, const __bf16* __restrict__ ctxbf,
                            const __bf16* __restrict__ wbf, const float* __restrict__ bias,
                            const float* __restrict__ mask, float* __restrict__ out) {
  const int mb = blockIdx.x; // 1024 blocks x 32 rows
  const int b = mb >> 5, q0 = (mb & 31) * 32;
  const int tid = threadIdx.x;
  const int w = tid >> 5, l = tid & 31;
  const int r = l & 15, h = l >> 4;
  const int qs = w & 1;

  const size_t mrow = (size_t)b * SDn + q0 + qs * 16;
  const __bf16* Ap = pbf + (mrow + r) * Dn;
  const __bf16* Ac = ctxbf + (mrow + r) * Dn;

  const __bf16* Bcs[4];
  int n0s[4];
#pragma unroll
  for (int t = 0; t < 4; ++t) {
    n0s[t] = ((w >> 1) + 4 * t) * 16;
    Bcs[t] = wbf + (size_t)(n0s[t] + r) * (2 * Dn);
  }

  v8f acc[4] = {};
#pragma unroll
  for (int c = 0; c < 8; ++c) {                       // K 0..255 from padded_seqs
    v16bf Afr = ldA(Ap, c * 32, h);
#pragma unroll
    for (int t = 0; t < 4; ++t)
      acc[t] = wmma_bf16(Afr, ldB(Bcs[t], c * 32, h), acc[t]);
  }
#pragma unroll
  for (int c = 0; c < 8; ++c) {                       // K 256..511 from context
    v16bf Afr = ldA(Ac, c * 32, h);
#pragma unroll
    for (int t = 0; t < 4; ++t)
      acc[t] = wmma_bf16(Afr, ldB(Bcs[t], 256 + c * 32, h), acc[t]);
  }

  float* orow = out + mrow * Dn;
  const float* mrw = mask + mrow;
#pragma unroll
  for (int t = 0; t < 4; ++t) {
    const float bv = bias[n0s[t] + r];
#pragma unroll
    for (int i = 0; i < 8; ++i) {
      float v = tanhf(acc[t][i] + bv) * mrw[h * 8 + i];
      orow[(size_t)(h * 8 + i) * Dn + n0s[t] + r] = v;
    }
  }
}

extern "C" void kernel_launch(void* const* d_in, const int* in_sizes, int n_in,
                              void* d_out, int out_size, void* d_ws, size_t ws_size,
                              hipStream_t stream) {
  (void)in_sizes; (void)n_in; (void)out_size; (void)ws_size;
  const float* p   = (const float*)d_in[0]; // (B, SD, D)
  const float* e   = (const float*)d_in[1]; // (B, SE, D)
  const float* msk = (const float*)d_in[2]; // (B, SD, 1)
  const float* W   = (const float*)d_in[3]; // (D, 2D)
  const float* bv  = (const float*)d_in[4]; // (D,)

  char* ws = (char*)d_ws;
  __bf16* pbf   = (__bf16*)(ws);              // 16 MiB
  __bf16* ebf   = (__bf16*)(ws + 16777216);   // 16 MiB  [b][e][d]
  __bf16* etbf  = (__bf16*)(ws + 33554432);   // 16 MiB  [b][d][e]
  __bf16* ctxbf = (__bf16*)(ws + 50331648);   // 16 MiB
  __bf16* wbf   = (__bf16*)(ws + 67108864);   // 256 KiB

  float* logits  = (float*)d_out;                  // (B, SD, D)
  float* weights = logits + (size_t)Bn * SDn * Dn; // (B, SD, SE)

  const size_t smem = (size_t)32 * SEn * 4 + (size_t)32 * SEn * 2 + 32 * 8 * 4;

  cvt_pw_kernel<<<2048, 256, 0, stream>>>(p, W, pbf, wbf);
  enc_cvt_tr_kernel<<<dim3(SEn / 32, Dn / 32, Bn), 256, 0, stream>>>(e, ebf, etbf);
  attn_kernel<<<dim3(SDn / 32, Bn), 256, smem, stream>>>(pbf, ebf, etbf, weights, ctxbf);
  proj_kernel<<<1024, 256, 0, stream>>>(pbf, ctxbf, wbf, bv, msk, logits);
}
